// Decoder_Ankle_81810537054561
// MI455X (gfx1250) — compile-verified, matrix operand order checked by reading, not run on hardware
//
#include <hip/hip_runtime.h>
#include <hip/hip_bf16.h>
#include <math.h>

#define Bsz 256
#define Ssz 256
#define Hsz 256

typedef __attribute__((ext_vector_type(16))) __bf16 v16bf;
typedef __attribute__((ext_vector_type(8)))  __bf16 v8bf;
typedef __attribute__((ext_vector_type(8)))  float  v8f;

// ---------------------------------------------------------------------------
// Fragment loader: 16-bit A/B matrix layout (ISA 7.12.2).
// Lane L (m = L&15, hi = L>>4) holds K = {hi*8 .. hi*8+7} and {16+hi*8 .. +7}
// => two contiguous 16-byte loads per lane from a row-major [rows, ld] tile.
// ---------------------------------------------------------------------------
__device__ __forceinline__ v16bf frag_load(const __bf16* __restrict__ p) {
    v8bf lo = *(const v8bf*)p;
    v8bf hi = *(const v8bf*)(p + 16);
    v16bf f;
#pragma unroll
    for (int e = 0; e < 8; ++e) { f[e] = lo[e]; f[e + 8] = hi[e]; }
    return f;
}

// ---------------------------------------------------------------------------
// Generic bf16 WMMA GEMM:  C[M,N] = A[M,K] @ Bt[N,K]^T  (+ bias[N])
// A row-major (lda), Bt stored transposed row-major [N,K] (ldb) so both
// fragments use the identical A-style layout. One wave owns a 16x64 strip
// (4 accumulators, A-fragment reused 4x). Output f32 (Cf) or bf16 (Cb).
// ---------------------------------------------------------------------------
__global__ void gemm_bf16_wmma(const __bf16* __restrict__ A, int lda,
                               const __bf16* __restrict__ Bt, int ldb,
                               float* __restrict__ Cf, __bf16* __restrict__ Cb,
                               const float* __restrict__ bias,
                               int M, int N, int K)
{
    int wid  = (blockIdx.x * blockDim.x + threadIdx.x) >> 5;
    int lane = threadIdx.x & 31;
    int lm = lane & 15, lh = lane >> 4;
    int ntn = N >> 6;                 // number of 64-wide n strips
    int tm  = wid / ntn;
    int tn  = wid % ntn;
    if (tm >= (M >> 4)) return;
    int m0 = tm << 4, n0 = tn << 6;

    v8f acc[4] = {};
    const __bf16* arow = A + (size_t)(m0 + lm) * lda + lh * 8;
    for (int k0 = 0; k0 < K; k0 += 32) {
        v16bf af = frag_load(arow + k0);
#pragma unroll
        for (int j = 0; j < 4; ++j) {
            const __bf16* bp = Bt + (size_t)(n0 + j * 16 + lm) * ldb + k0 + lh * 8;
            v16bf bf = frag_load(bp);
            acc[j] = __builtin_amdgcn_wmma_f32_16x16x32_bf16(
                         false, af, false, bf, (short)0, acc[j], false, false);
        }
    }
    // C/D layout: element r of lane L -> row m0 + r + 8*(L>>4), col n0 + (L&15)
#pragma unroll
    for (int j = 0; j < 4; ++j) {
        int col = n0 + j * 16 + lm;
        float bsv = bias ? bias[col] : 0.0f;
#pragma unroll
        for (int r = 0; r < 8; ++r) {
            int row = m0 + r + lh * 8;
            float v = acc[j][r] + bsv;
            if (Cb) Cb[(size_t)row * N + col] = (__bf16)v;
            else    Cf[(size_t)row * N + col] = v;
        }
    }
}

// ---------------------------------------------------------------------------
// Attention core: one workgroup (256 thr = 8 waves) per batch row b.
//  score[s] = b_sh + sum_k W_sh[k] * tanh(E[b,s,k] + hc[b,k])   (E has b_cat)
//  w = softmax(score); ctx[h] = sum_s w[s]*enc[b,s,h]
// E and enc are bf16 and L2-resident (32 MB each, 192 MB L2).
// ---------------------------------------------------------------------------
__global__ void attn_kernel(const __bf16* __restrict__ E,
                            const __bf16* __restrict__ encb,
                            const float* __restrict__ hcproj,
                            const float* __restrict__ W_sh,
                            const float* __restrict__ b_sh,
                            float* __restrict__ attn_out,
                            float* __restrict__ ctx_out, int t)
{
    __shared__ float hcL[Hsz], wshL[Hsz], sc[Ssz], red[Ssz];
    int b = blockIdx.x, tid = threadIdx.x;
    hcL[tid]  = hcproj[b * Hsz + tid];
    wshL[tid] = W_sh[tid];
    __syncthreads();

    int w = tid >> 5, lane = tid & 31;
    const __bf16* Eb = E + (size_t)b * Ssz * Hsz;
    for (int r = 0; r < 32; ++r) {               // 8 waves x 32 rows = 256 s
        int s = w * 32 + r;
        const __bf16* Es = Eb + s * Hsz;
        float a = 0.f;
#pragma unroll
        for (int kk = 0; kk < 8; ++kk) {         // 32 lanes x 8 = H
            int k = lane + kk * 32;
            a += wshL[k] * tanhf((float)Es[k] + hcL[k]);
        }
#pragma unroll
        for (int off = 16; off > 0; off >>= 1) a += __shfl_xor(a, off, 32);
        if (lane == 0) sc[s] = a + b_sh[0];
    }
    __syncthreads();

    // softmax over S within the block
    red[tid] = sc[tid]; __syncthreads();
    for (int off = 128; off > 0; off >>= 1) {
        if (tid < off) red[tid] = fmaxf(red[tid], red[tid + off]);
        __syncthreads();
    }
    float mx = red[0]; __syncthreads();
    float ev = expf(sc[tid] - mx);
    red[tid] = ev; __syncthreads();
    for (int off = 128; off > 0; off >>= 1) {
        if (tid < off) red[tid] += red[tid + off];
        __syncthreads();
    }
    float wgt = ev / red[0];
    sc[tid] = wgt;
    attn_out[((size_t)b * Ssz + t) * Ssz + tid] = wgt;
    __syncthreads();

    // ctx: thread tid owns channel h=tid; loads coalesced across threads
    const __bf16* eb = encb + (size_t)b * Ssz * Hsz + tid;
    float acc = 0.f;
    for (int s = 0; s < Ssz; ++s) acc += sc[s] * (float)eb[s * Hsz];
    ctx_out[((size_t)b * Ssz + t) * Hsz + tid] = acc;
}

// feeder: y_tilde[b] = b_feed + W_feed[0]*y_t[b] + sum_h W_feed[1+h]*ctx[b,h]
__global__ void feeder_kernel(const float* __restrict__ y_hist,
                              const float* __restrict__ ctx_all,
                              const float* __restrict__ W_feed,
                              const float* __restrict__ b_feed,
                              float* __restrict__ y_tilde, int t)
{
    int b = threadIdx.x;                          // 1 block x 256 threads
    float a = b_feed[0] + W_feed[0] * y_hist[b * Ssz + t];
    const float* cx = ctx_all + ((size_t)b * Ssz + t) * Hsz;
    for (int h = 0; h < Hsz; ++h) a += W_feed[1 + h] * cx[h];
    y_tilde[b] = a;
}

// LSTM pointwise epilogue: folds y_tilde*W_ih + (b_ih+b_hh) into the raw
// h@W_hh accumulator, updates c, and writes h/c (bf16) into the packed
// [B,512] hc buffer consumed by both next-step WMMA GEMMs.
__global__ void lstm_kernel(const float* __restrict__ G,
                            const float* __restrict__ W_ih,
                            const float* __restrict__ bihh,
                            const float* __restrict__ y_tilde,
                            float* __restrict__ c_state,
                            __bf16* __restrict__ hc_in,
                            float* __restrict__ dec_out, int t)
{
    int idx = blockIdx.x * blockDim.x + threadIdx.x;   // B*H
    if (idx >= Bsz * Hsz) return;
    int b = idx >> 8, h = idx & 255;
    float yt = y_tilde[b];
    const float* g = G + (size_t)b * 1024;
    float i_ = g[h]       + yt * W_ih[h]       + bihh[h];
    float f_ = g[256 + h] + yt * W_ih[256 + h] + bihh[256 + h];
    float gg = g[512 + h] + yt * W_ih[512 + h] + bihh[512 + h];
    float o_ = g[768 + h] + yt * W_ih[768 + h] + bihh[768 + h];
    float si = 1.f / (1.f + expf(-i_));
    float sf = 1.f / (1.f + expf(-f_));
    float so = 1.f / (1.f + expf(-o_));
    float cn = sf * c_state[idx] + si * tanhf(gg);
    float hn = so * tanhf(cn);
    c_state[idx] = cn;
    hc_in[(size_t)b * 512 + h]       = (__bf16)hn;
    hc_in[(size_t)b * 512 + 256 + h] = (__bf16)cn;
    dec_out[((size_t)b * Ssz + t) * Hsz + h] = hn;
}

// final fc: y[b] = ((concat[h_T, ctx_T] @ W_f1 + b_f1) @ W_f2) + b_f2
__global__ void final_fc_kernel(const float* __restrict__ dec_all,
                                const float* __restrict__ ctx_all,
                                const float* __restrict__ W_f1,
                                const float* __restrict__ b_f1,
                                const float* __restrict__ W_f2,
                                const float* __restrict__ b_f2,
                                float* __restrict__ y_out)
{
    __shared__ float red[Hsz];
    int b = blockIdx.x, tid = threadIdx.x;
    const float* hT = dec_all + ((size_t)b * Ssz + (Ssz - 1)) * Hsz;
    const float* cT = ctx_all + ((size_t)b * Ssz + (Ssz - 1)) * Hsz;
    float a = b_f1[tid];
    for (int j = 0; j < Hsz; ++j)
        a += hT[j] * W_f1[j * Hsz + tid] + cT[j] * W_f1[(Hsz + j) * Hsz + tid];
    red[tid] = a * W_f2[tid];
    __syncthreads();
    for (int off = 128; off > 0; off >>= 1) {
        if (tid < off) red[tid] += red[tid + off];
        __syncthreads();
    }
    if (tid == 0) y_out[b] = red[0] + b_f2[0];
}

// ---------------------------------------------------------------------------
// One-shot preparation kernels
// ---------------------------------------------------------------------------
__global__ void enc_to_bf16(const float* __restrict__ x, __bf16* __restrict__ y, int n)
{
    int i = blockIdx.x * blockDim.x + threadIdx.x;
    int st = gridDim.x * blockDim.x;
    for (; i < n; i += st) y[i] = (__bf16)x[i];
}

// Transpose + bf16-convert weights so WMMA B-fragments load like A-fragments.
__global__ void prep_weights(const float* __restrict__ W_cat,
                             const float* __restrict__ W_hh,
                             const float* __restrict__ b_ih,
                             const float* __restrict__ b_hh,
                             __bf16* __restrict__ WcatT_hc,   // [256 n, 512 k]
                             __bf16* __restrict__ WcatT_enc,  // [256 n, 256 k]
                             __bf16* __restrict__ WhhT,       // [1024 n, 256 k]
                             float* __restrict__ bihh)        // [1024]
{
    int i = blockIdx.x * blockDim.x + threadIdx.x;
    int st = gridDim.x * blockDim.x;
    for (; i < 262144; i += st) {
        int n = i >> 8, k = i & 255;
        WhhT[i] = (__bf16)W_hh[k * 1024 + n];
        if (i < 131072) { int n2 = i >> 9, k2 = i & 511; WcatT_hc[i]  = (__bf16)W_cat[k2 * 256 + n2]; }
        if (i < 65536)  { int n3 = i >> 8, k3 = i & 255; WcatT_enc[i] = (__bf16)W_cat[(512 + k3) * 256 + n3]; }
        if (i < 1024)   bihh[i] = b_ih[i] + b_hh[i];
    }
}

__global__ void init_state(__bf16* __restrict__ hc_in, float* __restrict__ c_state)
{
    int i = blockIdx.x * blockDim.x + threadIdx.x;
    if (i < Bsz * 512) hc_in[i] = (__bf16)0.f;
    if (i < Bsz * Hsz) c_state[i] = 0.f;
}

// ---------------------------------------------------------------------------
extern "C" void kernel_launch(void* const* d_in, const int* in_sizes, int n_in,
                              void* d_out, int out_size, void* d_ws, size_t ws_size,
                              hipStream_t stream)
{
    const float* enc    = (const float*)d_in[0];
    const float* y_hist = (const float*)d_in[1];
    const float* W_cat  = (const float*)d_in[2];
    const float* b_cat  = (const float*)d_in[3];
    const float* W_sh   = (const float*)d_in[4];
    const float* b_sh   = (const float*)d_in[5];
    const float* W_feed = (const float*)d_in[6];
    const float* b_feed = (const float*)d_in[7];
    const float* W_ih   = (const float*)d_in[8];
    const float* W_hh   = (const float*)d_in[9];
    const float* b_ih   = (const float*)d_in[10];
    const float* b_hh   = (const float*)d_in[11];
    const float* W_f1   = (const float*)d_in[12];
    const float* b_f1   = (const float*)d_in[13];
    const float* W_f2   = (const float*)d_in[14];
    const float* b_f2   = (const float*)d_in[15];

    const size_t BSH = (size_t)Bsz * Ssz * Hsz;      // 16,777,216
    float* y_out    = (float*)d_out;                 // [B,1]
    float* dec_all  = y_out + Bsz;                   // [B,S,H]
    float* ctx_all  = dec_all + BSH;                 // [B,S,H]
    float* attn_all = ctx_all + BSH;                 // [B,S,S]

    // workspace carve-up (~70 MB)
    char* ws = (char*)d_ws;
    size_t off = 0;
    auto carve = [&](size_t bytes) -> void* {
        void* p = ws + off;
        off = (off + bytes + 255) & ~(size_t)255;
        return p;
    };
    __bf16* encb      = (__bf16*)carve(BSH * 2);          // enc in bf16 (32 MB, L2-resident)
    __bf16* Eb        = (__bf16*)carve(BSH * 2);          // E = enc@W_e + b_cat (32 MB)
    __bf16* WcatT_hc  = (__bf16*)carve(256 * 512 * 2);
    __bf16* WcatT_enc = (__bf16*)carve(256 * 256 * 2);
    __bf16* WhhT      = (__bf16*)carve(1024 * 256 * 2);
    float*  bihh      = (float*) carve(1024 * 4);
    __bf16* hc_in     = (__bf16*)carve((size_t)Bsz * 512 * 2);  // packed [h|c] bf16
    float*  c_state   = (float*) carve((size_t)Bsz * Hsz * 4);
    float*  hcproj    = (float*) carve((size_t)Bsz * Hsz * 4);
    float*  Gbuf      = (float*) carve((size_t)Bsz * 1024 * 4);
    float*  y_tilde   = (float*) carve((size_t)Bsz * 4);
    (void)ws_size; (void)in_sizes; (void)n_in; (void)out_size;

    // --- one-shot prep ---
    enc_to_bf16 <<<8192, 256, 0, stream>>>(enc, encb, (int)BSH);
    prep_weights<<<256,  256, 0, stream>>>(W_cat, W_hh, b_ih, b_hh,
                                           WcatT_hc, WcatT_enc, WhhT, bihh);
    init_state  <<<512,  256, 0, stream>>>(hc_in, c_state);

    // Hoisted attention GEMM: E[BS,H] = encb[BS,H] @ W_e^T + b_cat  (bf16 out)
    // M=65536, N=256, K=256 -> 16384 wave-tiles -> 2048 blocks
    gemm_bf16_wmma<<<2048, 256, 0, stream>>>(encb, 256, WcatT_enc, 256,
                                             nullptr, Eb, b_cat, 65536, 256, 256);

    // --- sequential scan ---
    for (int t = 0; t < Ssz; ++t) {
        // hcproj = [h|c] @ W_cat[0:2H]   M=256,N=256,K=512 -> 8 blocks
        gemm_bf16_wmma<<<8, 256, 0, stream>>>(hc_in, 512, WcatT_hc, 512,
                                              hcproj, nullptr, nullptr, 256, 256, 512);
        attn_kernel  <<<Bsz, 256, 0, stream>>>(Eb, encb, hcproj, W_sh, b_sh,
                                               attn_all, ctx_all, t);
        feeder_kernel<<<1, 256, 0, stream>>>(y_hist, ctx_all, W_feed, b_feed,
                                             y_tilde, t);
        // gates = h @ W_hh              M=256,N=1024,K=256 -> 32 blocks
        gemm_bf16_wmma<<<32, 256, 0, stream>>>(hc_in, 512, WhhT, 256,
                                               Gbuf, nullptr, nullptr, 256, 1024, 256);
        lstm_kernel  <<<Bsz, 256, 0, stream>>>(Gbuf, W_ih, bihh, y_tilde,
                                               c_state, hc_in, dec_all, t);
    }

    final_fc_kernel<<<Bsz, 256, 0, stream>>>(dec_all, ctx_all, W_f1, b_f1,
                                             W_f2, b_f2, y_out);
}
